// ThreeLayerGAT_429496730273
// MI455X (gfx1250) — matrix operand order repaired and checked by reading, not compiled, not run on hardware
//
#include <hip/hip_runtime.h>
#include <hip/hip_bf16.h>
#include <stdint.h>

// ---------------- problem constants ----------------
#define NN      30000
#define EE      300000
#define ETOT    (EE + NN)      // with self loops
#define F_IN    128
#define HC12    1024           // 4 heads x 256
#define C12     256
#define H12     4
#define H3      6
#define C3      40
#define HC3     240
#define NEG_SLOPE 0.2f

typedef __attribute__((ext_vector_type(16))) __bf16 v16bf;
typedef __attribute__((ext_vector_type(8)))  __bf16 v8bf;
typedef __attribute__((ext_vector_type(8)))  float  v8f;
typedef __attribute__((ext_vector_type(4)))  float  v4f;

// =====================================================================
// Weight pre-swizzle: fp32 W[K,N] -> bf16 hi/lo in WMMA B-fragment order.
// Per 32x16 (KxN) tile: element (k,n) -> lane = (n&15) + ((k&31)>>4)*16,
//                                        e    = k & 15
// Tile storage: 512 halfs, [lane][e] contiguous (32B per lane, 16B aligned).
// =====================================================================
__global__ void k_swizzleW(const float* __restrict__ W, __bf16* __restrict__ Bh,
                           __bf16* __restrict__ Bl, int K, int N)
{
    int t = blockIdx.x * 256 + threadIdx.x;
    if (t >= K * N) return;
    int k = t / N, n = t % N;
    float v = W[t];
    __bf16 h = (__bf16)v;
    __bf16 l = (__bf16)(v - (float)h);
    int lane = (n & 15) + (((k >> 4) & 1) << 4);
    int e    = k & 15;
    size_t idx = ((size_t)((k >> 5) * (N >> 4) + (n >> 4)) << 9) + lane * 16 + e;
    Bh[idx] = h;
    Bl[idx] = l;
}

// =====================================================================
// bf16x3 WMMA GEMM:  C[M,N] = A[M,K] * B[K,N]   (fp32 in/out, fp32 acc)
// Block tile 128x128, K-tile 32, 256 threads = 8 waves (4x2 wave grid),
// wave tile 32x64 = 2x4 fragments of 16x16.
// A: fp32 global -> fragment-major LDS (v8bf stores / v16bf loads).
// B: pre-swizzled bf16 fragments loaded straight from global (L2-resident).
// =====================================================================
__global__ __launch_bounds__(256)
void gat_gemm_bf16x3(const float* __restrict__ A,
                     const __bf16* __restrict__ Bh, const __bf16* __restrict__ Bl,
                     float* __restrict__ C, int M, int N, int K)
{
    // [m-tile][lane][e] : per-lane fragment rows are 32B and 32B-aligned
    __shared__ __bf16 Ah[8][32][16];
    __shared__ __bf16 Al[8][32][16];

    const int t    = threadIdx.x;
    const int m0   = blockIdx.y * 128;
    const int n0   = blockIdx.x * 128;
    const int wave = t >> 5;
    const int lane = t & 31;
    const int half = lane >> 4;
    const int l16  = lane & 15;
    const int wm   = (wave >> 1) * 32;   // wave M offset in block tile
    const int wn   = (wave & 1) * 64;    // wave N offset in block tile
    const int Nt   = N >> 4;             // # 16-wide n-tiles
    const int Kt   = K >> 5;             // # 32-wide k-tiles

    // A loader assignment: group g covers 8 consecutive k of one row
    // k-group g: half = g>>1, e0 = (g&1)*8  (matches A 16x32 fragment map)
    const int g     = t & 3;
    const int ahalf = g >> 1;
    const int ae0   = (g & 1) * 8;

    v8f acc[2][4];
#pragma unroll
    for (int mi = 0; mi < 2; ++mi)
#pragma unroll
        for (int ni = 0; ni < 4; ++ni)
#pragma unroll
            for (int r = 0; r < 8; ++r) acc[mi][ni][r] = 0.f;

    for (int ks = 0; ks < Kt; ++ks) {
        const int k0 = ks << 5;
        __syncthreads();
        // ---- fp32 global -> (hi,lo) bf16 fragment-major LDS ----
#pragma unroll
        for (int i = 0; i < 2; ++i) {
            int r  = (t >> 2) + i * 64;
            int gr = m0 + r;
            v4f f0 = {0.f, 0.f, 0.f, 0.f};
            v4f f1 = {0.f, 0.f, 0.f, 0.f};
            if (gr < M) {
                const float* ap = A + (size_t)gr * K + (k0 + g * 8);
                f0 = *(const v4f*)ap;
                f1 = *(const v4f*)(ap + 4);
            }
            v8bf hi8, lo8;
#pragma unroll
            for (int j = 0; j < 4; ++j) {
                float v  = f0[j];
                __bf16 h = (__bf16)v;
                hi8[j]   = h;
                lo8[j]   = (__bf16)(v - (float)h);
            }
#pragma unroll
            for (int j = 0; j < 4; ++j) {
                float v    = f1[j];
                __bf16 h   = (__bf16)v;
                hi8[4 + j] = h;
                lo8[4 + j] = (__bf16)(v - (float)h);
            }
            int alane = (r & 15) + ahalf * 16;
            int mt    = r >> 4;
            *(v8bf*)&Ah[mt][alane][ae0] = hi8;
            *(v8bf*)&Al[mt][alane][ae0] = lo8;
        }
        __syncthreads();

        // ---- fragment loads: contiguous 32B per lane ----
        v16bf a_hi[2], a_lo[2];
#pragma unroll
        for (int mi = 0; mi < 2; ++mi) {
            int mt    = (wm >> 4) + mi;
            a_hi[mi]  = *(const v16bf*)&Ah[mt][lane][0];
            a_lo[mi]  = *(const v16bf*)&Al[mt][lane][0];
        }
        v16bf b_hi[4], b_lo[4];
#pragma unroll
        for (int ni = 0; ni < 4; ++ni) {
            int nt = (n0 >> 4) + (wn >> 4) + ni;
            if (nt < Nt) {
                size_t base = ((size_t)(ks * Nt + nt) << 9) + lane * 16;
                b_hi[ni] = *(const v16bf*)(Bh + base);
                b_lo[ni] = *(const v16bf*)(Bl + base);
            } else {
#pragma unroll
                for (int e = 0; e < 16; ++e) { b_hi[ni][e] = (__bf16)0.f; b_lo[ni][e] = (__bf16)0.f; }
            }
        }

        // ---- bf16x3: hi*hi + hi*lo + lo*hi, fp32 accumulate ----
#pragma unroll
        for (int mi = 0; mi < 2; ++mi)
#pragma unroll
            for (int ni = 0; ni < 4; ++ni) {
                acc[mi][ni] = __builtin_amdgcn_wmma_f32_16x16x32_bf16(
                    false, a_hi[mi], false, b_hi[ni], (short)0, acc[mi][ni], false, false);
                acc[mi][ni] = __builtin_amdgcn_wmma_f32_16x16x32_bf16(
                    false, a_hi[mi], false, b_lo[ni], (short)0, acc[mi][ni], false, false);
                acc[mi][ni] = __builtin_amdgcn_wmma_f32_16x16x32_bf16(
                    false, a_lo[mi], false, b_hi[ni], (short)0, acc[mi][ni], false, false);
            }
    }

    // ---- epilogue: C/D layout row = r + half*8, col = l16 ----
#pragma unroll
    for (int mi = 0; mi < 2; ++mi)
#pragma unroll
        for (int ni = 0; ni < 4; ++ni)
#pragma unroll
            for (int r = 0; r < 8; ++r) {
                int row = m0 + wm + mi * 16 + r + half * 8;
                int col = n0 + wn + ni * 16 + l16;
                if (row < M && col < N) C[(size_t)row * N + col] = acc[mi][ni][r];
            }
}

// =====================================================================
// Edge preparation and CSR-by-dst construction (built once, reused x3)
// =====================================================================
__global__ void k_prep_edges(const long long* __restrict__ eidx,
                             int* __restrict__ src32, int* __restrict__ dst32)
{
    int e = blockIdx.x * 256 + threadIdx.x;
    if (e >= ETOT) return;
    if (e < EE) { src32[e] = (int)eidx[e]; dst32[e] = (int)eidx[EE + e]; }
    else        { src32[e] = dst32[e] = e - EE; }               // self loops
}

__global__ void k_count(const int* __restrict__ dst32, int* __restrict__ counts)
{
    int e = blockIdx.x * 256 + threadIdx.x;
    if (e >= ETOT) return;
    atomicAdd(&counts[dst32[e]], 1);
}

__global__ __launch_bounds__(1024)
void k_scan(const int* __restrict__ counts, int* __restrict__ offs, int n)
{
    __shared__ int tmp[1024];
    __shared__ int carry_s;
    int tid = threadIdx.x;
    if (tid == 0) carry_s = 0;
    __syncthreads();
    for (int base = 0; base < n; base += 1024) {
        int v = (base + tid < n) ? counts[base + tid] : 0;
        tmp[tid] = v;
        __syncthreads();
        for (int off = 1; off < 1024; off <<= 1) {
            int u = (tid >= off) ? tmp[tid - off] : 0;
            __syncthreads();
            tmp[tid] += u;
            __syncthreads();
        }
        int carry = carry_s;
        if (base + tid < n) offs[base + tid] = carry + tmp[tid] - v;  // exclusive
        __syncthreads();
        if (tid == 1023) carry_s = carry + tmp[1023];
        __syncthreads();
    }
    if (tid == 0) offs[n] = carry_s;
}

__global__ void k_fill(const int* __restrict__ dst32, const int* __restrict__ row_off,
                       int* __restrict__ cursor, int* __restrict__ csr_eid)
{
    int e = blockIdx.x * 256 + threadIdx.x;
    if (e >= ETOT) return;
    int d = dst32[e];
    int pos = row_off[d] + atomicAdd(&cursor[d], 1);
    csr_eid[pos] = e;
}

// =====================================================================
// Attention pieces
// =====================================================================
// es[n,h] = sum_c h[n,h,c]*a_src[h,c] ; ed likewise. One wave per (n,h).
__global__ void k_node_logits(const float* __restrict__ Hpre,
                              const float* __restrict__ a_src,
                              const float* __restrict__ a_dst,
                              float* __restrict__ es, float* __restrict__ ed,
                              int H, int Cc)
{
    int wave = (blockIdx.x * 256 + threadIdx.x) >> 5;
    int lane = threadIdx.x & 31;
    if (wave >= NN * H) return;
    int n = wave / H, h = wave % H;
    const float* hp  = Hpre + (size_t)n * H * Cc + (size_t)h * Cc;
    const float* asv = a_src + (size_t)h * Cc;
    const float* adv = a_dst + (size_t)h * Cc;
    float s0 = 0.f, s1 = 0.f;
    for (int c = lane; c < Cc; c += 32) {
        float v = hp[c];
        s0 += v * asv[c];
        s1 += v * adv[c];
    }
#pragma unroll
    for (int off = 16; off > 0; off >>= 1) {
        s0 += __shfl_xor(s0, off, 32);
        s1 += __shfl_xor(s1, off, 32);
    }
    if (lane == 0) { es[wave] = s0; ed[wave] = s1; }
}

// e = leaky_relu(es[src]+ed[dst]) per edge/head
__global__ void k_edge_logits(const float* __restrict__ es, const float* __restrict__ ed,
                              const int* __restrict__ src32, const int* __restrict__ dst32,
                              float* __restrict__ e_buf, int H)
{
    int e = blockIdx.x * 256 + threadIdx.x;
    if (e >= ETOT) return;
    int s = src32[e], d = dst32[e];
    for (int h = 0; h < H; ++h) {
        float v = es[s * H + h] + ed[d * H + h];
        e_buf[(size_t)e * H + h] = (v > 0.f) ? v : NEG_SLOPE * v;
    }
}

// Segment softmax via CSR: one thread per (node, head). No float atomics.
__global__ void k_softmax(const float* __restrict__ e_buf,
                          const int* __restrict__ row_off, const int* __restrict__ csr_eid,
                          float* __restrict__ alpha, float* __restrict__ alpha_out,
                          int H)
{
    int t = blockIdx.x * 256 + threadIdx.x;
    if (t >= NN * H) return;
    int n = t / H, h = t % H;
    int rs = row_off[n], re = row_off[n + 1];
    float m = -3.4e38f;
    for (int k = rs; k < re; ++k)
        m = fmaxf(m, e_buf[(size_t)csr_eid[k] * H + h]);
    float ssum = 0.f;
    for (int k = rs; k < re; ++k) {
        size_t idx = (size_t)csr_eid[k] * H + h;
        float ex = expf(e_buf[idx] - m);
        alpha[idx] = ex;
        ssum += ex;
    }
    float inv = 1.f / (ssum + 1e-16f);
    for (int k = rs; k < re; ++k) {
        size_t idx = (size_t)csr_eid[k] * H + h;
        float a = alpha[idx] * inv;
        alpha[idx] = a;
        if (alpha_out) alpha_out[idx] = a;
    }
}

// Gather aggregation: block per node; out[n,ch] = sum_e alpha[e,hd(ch)]*H[src,ch]
__global__ __launch_bounds__(256)
void k_aggregate(const float* __restrict__ Hpre, const float* __restrict__ alpha,
                 const int* __restrict__ row_off, const int* __restrict__ csr_eid,
                 const int* __restrict__ src32, const float* __restrict__ bias,
                 float* __restrict__ out, int HC, int Cc, int H, int do_elu)
{
    int n  = blockIdx.x;
    int rs = row_off[n], re = row_off[n + 1];
    int chs[4], hds[4];
#pragma unroll
    for (int j = 0; j < 4; ++j) {
        chs[j] = threadIdx.x + j * 256;
        hds[j] = (chs[j] < HC) ? (chs[j] / Cc) : 0;
    }
    float accv[4] = {0.f, 0.f, 0.f, 0.f};
    for (int k = rs; k < re; ++k) {
        int e = csr_eid[k];
        int s = src32[e];
        const float* hs = Hpre + (size_t)s * HC;
        const float* al = alpha + (size_t)e * H;
#pragma unroll
        for (int j = 0; j < 4; ++j)
            if (chs[j] < HC) accv[j] += al[hds[j]] * hs[chs[j]];
    }
#pragma unroll
    for (int j = 0; j < 4; ++j) {
        int ch = chs[j];
        if (ch < HC) {
            float v = accv[j] + (bias ? bias[ch] : 0.f);
            if (do_elu) v = (v > 0.f) ? v : (expf(v) - 1.f);
            out[(size_t)n * HC + ch] = v;
        }
    }
}

// Layer 3 head mean + bias
__global__ void k_mean_heads(const float* __restrict__ h3, const float* __restrict__ b3,
                             float* __restrict__ out)
{
    int t = blockIdx.x * 256 + threadIdx.x;
    if (t >= NN * C3) return;
    int n = t / C3, c = t % C3;
    float s = 0.f;
    for (int h = 0; h < H3; ++h) s += h3[(size_t)n * HC3 + h * C3 + c];
    out[(size_t)n * C3 + c] = s * (1.f / (float)H3) + b3[c];
}

// =====================================================================
extern "C" void kernel_launch(void* const* d_in, const int* in_sizes, int n_in,
                              void* d_out, int out_size, void* d_ws, size_t ws_size,
                              hipStream_t stream)
{
    const float*     x      = (const float*)d_in[0];
    const long long* eidx   = (const long long*)d_in[1];
    const float*     W1     = (const float*)d_in[2];
    const float*     a_src1 = (const float*)d_in[3];
    const float*     a_dst1 = (const float*)d_in[4];
    const float*     b1     = (const float*)d_in[5];
    const float*     W2     = (const float*)d_in[6];
    const float*     a_src2 = (const float*)d_in[7];
    const float*     a_dst2 = (const float*)d_in[8];
    const float*     b2     = (const float*)d_in[9];
    const float*     W3     = (const float*)d_in[10];
    const float*     a_src3 = (const float*)d_in[11];
    const float*     a_dst3 = (const float*)d_in[12];
    const float*     b3     = (const float*)d_in[13];

    float* out_h3    = (float*)d_out;                   // [30000, 40]
    float* out_alpha = (float*)d_out + (size_t)NN * C3; // [330000, 4]

    // ---- carve workspace ----
    char* base = (char*)d_ws;
    size_t off = 0;
    auto alloc = [&](size_t bytes) -> char* {
        off = (off + 255) & ~(size_t)255;
        char* p = base + off;
        off += bytes;
        return p;
    };
    float*  bufA   = (float*) alloc((size_t)NN * HC12 * sizeof(float)); // 122.9 MB
    float*  bufB   = (float*) alloc((size_t)NN * HC12 * sizeof(float)); // 122.9 MB
    __bf16* W1h    = (__bf16*)alloc((size_t)F_IN * HC12 * sizeof(__bf16));
    __bf16* W1l    = (__bf16*)alloc((size_t)F_IN * HC12 * sizeof(__bf16));
    __bf16* W2h    = (__bf16*)alloc((size_t)HC12 * HC12 * sizeof(__bf16));
    __bf16* W2l    = (__bf16*)alloc((size_t)HC12 * HC12 * sizeof(__bf16));
    __bf16* W3h    = (__bf16*)alloc((size_t)HC12 * HC3 * sizeof(__bf16));
    __bf16* W3l    = (__bf16*)alloc((size_t)HC12 * HC3 * sizeof(__bf16));
    int*    src32  = (int*)   alloc((size_t)ETOT * sizeof(int));
    int*    dst32  = (int*)   alloc((size_t)ETOT * sizeof(int));
    int*    counts = (int*)   alloc((size_t)NN * sizeof(int));
    int*    cursor = (int*)   alloc((size_t)NN * sizeof(int));
    int*    row_off= (int*)   alloc((size_t)(NN + 1) * sizeof(int));
    int*    csr    = (int*)   alloc((size_t)ETOT * sizeof(int));
    float*  es     = (float*) alloc((size_t)NN * H3 * sizeof(float));
    float*  ed     = (float*) alloc((size_t)NN * H3 * sizeof(float));
    float*  e_buf  = (float*) alloc((size_t)ETOT * H3 * sizeof(float));
    float*  alpha  = (float*) alloc((size_t)ETOT * H3 * sizeof(float));

    const int EB = (ETOT + 255) / 256;

    // ---- one-time: weight swizzles + CSR by dst ----
    k_swizzleW<<<(F_IN * HC12 + 255) / 256, 256, 0, stream>>>(W1, W1h, W1l, F_IN, HC12);
    k_swizzleW<<<(HC12 * HC12 + 255) / 256, 256, 0, stream>>>(W2, W2h, W2l, HC12, HC12);
    k_swizzleW<<<(HC12 * HC3  + 255) / 256, 256, 0, stream>>>(W3, W3h, W3l, HC12, HC3);

    k_prep_edges<<<EB, 256, 0, stream>>>(eidx, src32, dst32);
    hipMemsetAsync(counts, 0, (size_t)NN * sizeof(int), stream);
    hipMemsetAsync(cursor, 0, (size_t)NN * sizeof(int), stream);
    k_count<<<EB, 256, 0, stream>>>(dst32, counts);
    k_scan<<<1, 1024, 0, stream>>>(counts, row_off, NN);
    k_fill<<<EB, 256, 0, stream>>>(dst32, row_off, cursor, csr);

    const dim3 g_gemm12((HC12 + 127) / 128, (NN + 127) / 128);
    const dim3 g_gemm3 ((HC3  + 127) / 128, (NN + 127) / 128);
    const int NH12_blk = (NN * H12 + 7) / 8;   // 1 wave per (n,h), 8 waves/block
    const int NH3_blk  = (NN * H3 + 7) / 8;
    const int NT12_blk = (NN * H12 + 255) / 256;
    const int NT3_blk  = (NN * H3 + 255) / 256;

    // ---- Layer 1: x[30000,128] @ W1 -> 4 heads x 256, concat, ELU ----
    gat_gemm_bf16x3<<<g_gemm12, 256, 0, stream>>>(x, W1h, W1l, bufA, NN, HC12, F_IN);
    k_node_logits<<<NH12_blk, 256, 0, stream>>>(bufA, a_src1, a_dst1, es, ed, H12, C12);
    k_edge_logits<<<EB, 256, 0, stream>>>(es, ed, src32, dst32, e_buf, H12);
    k_softmax<<<NT12_blk, 256, 0, stream>>>(e_buf, row_off, csr, alpha, out_alpha, H12);
    k_aggregate<<<NN, 256, 0, stream>>>(bufA, alpha, row_off, csr, src32, b1, bufB,
                                        HC12, C12, H12, /*elu=*/1);

    // ---- Layer 2: h1[30000,1024] @ W2 -> 4 heads x 256, concat, ELU ----
    gat_gemm_bf16x3<<<g_gemm12, 256, 0, stream>>>(bufB, W2h, W2l, bufA, NN, HC12, HC12);
    k_node_logits<<<NH12_blk, 256, 0, stream>>>(bufA, a_src2, a_dst2, es, ed, H12, C12);
    k_edge_logits<<<EB, 256, 0, stream>>>(es, ed, src32, dst32, e_buf, H12);
    k_softmax<<<NT12_blk, 256, 0, stream>>>(e_buf, row_off, csr, alpha, nullptr, H12);
    k_aggregate<<<NN, 256, 0, stream>>>(bufA, alpha, row_off, csr, src32, b2, bufB,
                                        HC12, C12, H12, /*elu=*/1);

    // ---- Layer 3: h2[30000,1024] @ W3 -> 6 heads x 40, mean over heads ----
    gat_gemm_bf16x3<<<g_gemm3, 256, 0, stream>>>(bufB, W3h, W3l, bufA, NN, HC3, HC12);
    k_node_logits<<<NH3_blk, 256, 0, stream>>>(bufA, a_src3, a_dst3, es, ed, H3, C3);
    k_edge_logits<<<EB, 256, 0, stream>>>(es, ed, src32, dst32, e_buf, H3);
    k_softmax<<<NT3_blk, 256, 0, stream>>>(e_buf, row_off, csr, alpha, nullptr, H3);
    // aggregate into bufB (h2 no longer needed after GEMM3), no bias/elu yet
    k_aggregate<<<NN, 256, 0, stream>>>(bufA, alpha, row_off, csr, src32, nullptr, bufB,
                                        HC3, C3, H3, /*elu=*/0);
    k_mean_heads<<<(NN * C3 + 255) / 256, 256, 0, stream>>>(bufB, b3, out_h3);
}
